// TinyMambaMulti_44848048505422
// MI455X (gfx1250) — compile-verified, matrix operand order checked by reading, not run on hardware
//
#include <hip/hip_runtime.h>
#include <hip/hip_bf16.h>
#include <math.h>

// ---------------- model constants ----------------
#define BATCH     4
#define SEQ       4096
#define ROWS      (BATCH * SEQ)      // 16384 tokens
#define INPUT_DIM 57
#define KPAD_IN   64                 // x / w_in padded K
#define D_MODEL   64
#define D_INNER   128
#define D_STATE   16
#define D_CONV    4
#define DT_RANK   4
#define DBL_DIM   (DT_RANK + 2 * D_STATE)   // 36
#define N_LAYERS  2
#define OUT_SIZE  6
// chunked scan: 64 chunks x 64 steps
#define NC        64
#define TC        (SEQ / NC)

// ---------------- WMMA types ----------------
typedef __attribute__((ext_vector_type(16))) _Float16 v16h;
typedef __attribute__((ext_vector_type(8)))  _Float16 v8h;
typedef __attribute__((ext_vector_type(8)))  float    v8f;

__device__ __forceinline__ v8f wmma_f16(v16h a, v16h b, v8f c) {
  // D = A(16x32 f16) * B(32x16 f16) + C(16x16 f32)   [v_wmma_f32_16x16x32_f16]
  return __builtin_amdgcn_wmma_f32_16x16x32_f16(
      /*neg_a=*/false, a, /*neg_b=*/false, b,
      /*c_mod=*/(short)0, c, /*reuse_a=*/false, /*reuse_b=*/false);
}

// Fragment load from pre-packed f16 row-major [*, ld] (ld multiple of 32,
// 16B-aligned rows). Serves as A (r0 = M tile row) or B (r0 = N tile row of
// W[N,K], since B[k][n] = W[n][k]).  ISA 7.12.2 lane layout:
//   lane = (m = lane&15, h = lane>>4); elem i<8 : K = k0+8h+i ;
//   elem i>=8 : K = k0+16+8h+(i-8)  -> two contiguous 16B groups.
__device__ __forceinline__ v16h load_frag(const _Float16* __restrict__ P,
                                          int r0, int ld, int k0, int lane) {
  const int m = lane & 15, hh = lane >> 4;
  const _Float16* rp = P + (size_t)(r0 + m) * ld + k0 + hh * 8;
  const v8h lo = *(const v8h*)rp;          // K = k0+8h   .. +7
  const v8h hi = *(const v8h*)(rp + 16);   // K = k0+16+8h.. +7
  v16h v;
#pragma unroll
  for (int i = 0; i < 8; ++i) { v[i] = lo[i]; v[i + 8] = hi[i]; }
  return v;
}

// D/C fragment (16x16 f32): VGPR r at lane (n = lane&15, h = lane>>4) is
// element [M = r + 8h][N = n].
__device__ __forceinline__ v8f load_c(const float* __restrict__ C,
                                      int row0, int n0, int ld, int lane) {
  const int n = lane & 15, hh = lane >> 4;
  v8f acc;
#pragma unroll
  for (int r = 0; r < 8; ++r)
    acc[r] = C[(size_t)(row0 + r + 8 * hh) * ld + n0 + n];
  return acc;
}

__device__ __forceinline__ float silu_f(float v) {
  return v * (1.f / (1.f + __expf(-v)));
}

// ---------------- prep: f32 -> f16 with optional K padding ----------------
__global__ __launch_bounds__(256) void k_cvt_pad(
    _Float16* __restrict__ dst, const float* __restrict__ src,
    int N, int K, int Kpad) {
  const int idx = blockIdx.x * 256 + threadIdx.x;
  if (idx >= N * Kpad) return;
  const int n = idx / Kpad, k = idx - n * Kpad;
  dst[idx] = (k < K) ? (_Float16)src[(size_t)n * K + k] : (_Float16)0.f;
}

// ---------------- kernels ----------------

// h[16384,64] = xh[16384,64p] @ w_in_h^T + b_in ; writes f32 h + f16 twin hh
__global__ __launch_bounds__(128) void k_linear_in(
    const _Float16* __restrict__ xh, const _Float16* __restrict__ w_in_h,
    const float* __restrict__ b_in, float* __restrict__ h,
    _Float16* __restrict__ hh) {
  const int lane = threadIdx.x & 31, w = threadIdx.x >> 5;
  const int row0 = blockIdx.x * 16, n0 = w * 16;
  v8f acc = {};
#pragma unroll
  for (int k0 = 0; k0 < KPAD_IN; k0 += 32) {
    v16h a = load_frag(xh, row0, KPAD_IN, k0, lane);
    v16h b = load_frag(w_in_h, n0, KPAD_IN, k0, lane);
    acc = wmma_f16(a, b, acc);
  }
  const int n = lane & 15, hhf = lane >> 4;
  const float bias = b_in[n0 + n];
#pragma unroll
  for (int r = 0; r < 8; ++r) {
    const float v = acc[r] + bias;
    const size_t o = (size_t)(row0 + r + 8 * hhf) * D_MODEL + n0 + n;
    h[o] = v;
    hh[o] = (_Float16)v;
  }
}

// xz[16384,256] = hh[16384,64] @ ipw_h^T ; 8 waves x 2 N-tiles (A reused)
__global__ __launch_bounds__(256) void k_in_proj(
    const _Float16* __restrict__ hh, const _Float16* __restrict__ ipw_h,
    float* __restrict__ xz) {
  const int lane = threadIdx.x & 31, w = threadIdx.x >> 5;
  const int row0 = blockIdx.x * 16;
  const int n0a = (w * 2) * 16, n0b = n0a + 16;
  v8f acc0 = {}, acc1 = {};
#pragma unroll
  for (int k0 = 0; k0 < D_MODEL; k0 += 32) {
    v16h a  = load_frag(hh, row0, D_MODEL, k0, lane);
    v16h b0 = load_frag(ipw_h, n0a, D_MODEL, k0, lane);
    v16h b1 = load_frag(ipw_h, n0b, D_MODEL, k0, lane);
    acc0 = wmma_f16(a, b0, acc0);
    acc1 = wmma_f16(a, b1, acc1);
  }
  const int n = lane & 15, hhf = lane >> 4;
#pragma unroll
  for (int r = 0; r < 8; ++r) {
    const size_t ro = (size_t)(row0 + r + 8 * hhf) * (2 * D_INNER);
    xz[ro + n0a + n] = acc0[r];
    xz[ro + n0b + n] = acc1[r];
  }
}

// Fused: depthwise causal conv4 + SiLU -> u ; dbl = u @ xpw^T (36) ;
// delta = softplus(dt @ dpw^T + dpb). One block per token (r = b*SEQ + t).
__global__ __launch_bounds__(128) void k_mix(
    const float* __restrict__ xz, const float* __restrict__ cw,
    const float* __restrict__ cb, const float* __restrict__ xpw,
    const float* __restrict__ dpw, const float* __restrict__ dpb,
    float* __restrict__ u, float* __restrict__ delta,
    float* __restrict__ Bm, float* __restrict__ Cm) {
  const int r = blockIdx.x;
  const int t = r & (SEQ - 1);
  const int d = threadIdx.x;
  __shared__ float ush[D_INNER];
  __shared__ float dbl[DBL_DIM];

  float acc = cb[d];
  if (t >= D_CONV - 1) {            // block-uniform fast path (no divergence)
#pragma unroll
    for (int j = 0; j < D_CONV; ++j)
      acc = fmaf(cw[d * D_CONV + j],
                 xz[(size_t)(r - (D_CONV - 1) + j) * (2 * D_INNER) + d], acc);
  } else {
#pragma unroll
    for (int j = 0; j < D_CONV; ++j) {
      const int tp = t - (D_CONV - 1) + j;
      const float xv = (tp >= 0)
          ? xz[(size_t)(r - (D_CONV - 1) + j) * (2 * D_INNER) + d] : 0.f;
      acc = fmaf(cw[d * D_CONV + j], xv, acc);
    }
  }
  const float uv = silu_f(acc);
  ush[d] = uv;
  u[(size_t)r * D_INNER + d] = uv;
  __syncthreads();

  if (d < DBL_DIM) {
    const float* wrow = xpw + d * D_INNER;
    float s = 0.f;
#pragma unroll 8
    for (int k = 0; k < D_INNER; ++k) s = fmaf(wrow[k], ush[k], s);
    dbl[d] = s;
    if (d >= DT_RANK && d < DT_RANK + D_STATE)
      Bm[(size_t)r * D_STATE + (d - DT_RANK)] = s;
    else if (d >= DT_RANK + D_STATE)
      Cm[(size_t)r * D_STATE + (d - DT_RANK - D_STATE)] = s;
  }
  __syncthreads();

  float dv = dpb[d];
#pragma unroll
  for (int rr = 0; rr < DT_RANK; ++rr) dv = fmaf(dpw[d * DT_RANK + rr], dbl[rr], dv);
  const float sp = (dv > 20.f) ? dv : log1pf(__expf(dv));
  delta[(size_t)r * D_INNER + d] = sp;
}

// Scan pass 1: per (b, chunk, d) compute per-chunk decay product + end state.
__global__ __launch_bounds__(128) void k_scan1(
    const float* __restrict__ delta, const float* __restrict__ u,
    const float* __restrict__ Bm, const float* __restrict__ Alog,
    float* __restrict__ send, float* __restrict__ aprod) {
  const int c = blockIdx.x, b = blockIdx.y, d = threadIdx.x;
  __shared__ float Bsh[TC * D_STATE];
  for (int i = threadIdx.x; i < TC * D_STATE; i += 128)
    Bsh[i] = Bm[(size_t)(b * SEQ + c * TC + (i >> 4)) * D_STATE + (i & 15)];
  __syncthreads();

  float Ad[D_STATE];
#pragma unroll
  for (int n = 0; n < D_STATE; ++n) Ad[n] = -__expf(Alog[d * D_STATE + n]);
  float s[D_STATE], ap[D_STATE];
#pragma unroll
  for (int n = 0; n < D_STATE; ++n) { s[n] = 0.f; ap[n] = 1.f; }

  const size_t base = (size_t)(b * SEQ + c * TC) * D_INNER + d;
  for (int tt = 0; tt < TC; ++tt) {
    const float dl = delta[base + (size_t)tt * D_INNER];
    const float du = dl * u[base + (size_t)tt * D_INNER];
    const float* Bt = &Bsh[tt * D_STATE];
#pragma unroll
    for (int n = 0; n < D_STATE; ++n) {
      const float a = __expf(dl * Ad[n]);
      ap[n] *= a;
      s[n] = fmaf(a, s[n], du * Bt[n]);
    }
  }
  const size_t o = ((size_t)(b * NC + c) * D_INNER + d) * D_STATE;
#pragma unroll
  for (int n = 0; n < D_STATE; ++n) { send[o + n] = s[n]; aprod[o + n] = ap[n]; }
}

// Sequential combine across 64 chunks (8192 independent scalar recurrences).
__global__ __launch_bounds__(256) void k_combine(
    const float* __restrict__ send, const float* __restrict__ aprod,
    float* __restrict__ sinit) {
  const int e = blockIdx.x * 256 + threadIdx.x;   // 4*128*16 = 8192
  if (e >= BATCH * D_INNER * D_STATE) return;
  const int b = e / (D_INNER * D_STATE);
  const int dn = e % (D_INNER * D_STATE);
  float carry = 0.f;
  for (int c = 0; c < NC; ++c) {
    const size_t idx = (size_t)(b * NC + c) * (D_INNER * D_STATE) + dn;
    sinit[idx] = carry;
    carry = fmaf(aprod[idx], carry, send[idx]);
  }
}

// Scan pass 2: re-run chunk with carry-in, emit y = (scan + u*D) * silu(z)
// directly in f16 (only consumed by out_proj WMMA).
__global__ __launch_bounds__(128) void k_scan2(
    const float* __restrict__ delta, const float* __restrict__ u,
    const float* __restrict__ Bm, const float* __restrict__ Cm,
    const float* __restrict__ Alog, const float* __restrict__ Dskip,
    const float* __restrict__ xz, const float* __restrict__ sinit,
    _Float16* __restrict__ yh) {
  const int c = blockIdx.x, b = blockIdx.y, d = threadIdx.x;
  __shared__ float Bsh[TC * D_STATE];
  __shared__ float Csh[TC * D_STATE];
  for (int i = threadIdx.x; i < TC * D_STATE; i += 128) {
    const size_t g = (size_t)(b * SEQ + c * TC + (i >> 4)) * D_STATE + (i & 15);
    Bsh[i] = Bm[g];
    Csh[i] = Cm[g];
  }
  __syncthreads();

  float Ad[D_STATE];
#pragma unroll
  for (int n = 0; n < D_STATE; ++n) Ad[n] = -__expf(Alog[d * D_STATE + n]);
  float s[D_STATE];
  const size_t so = ((size_t)(b * NC + c) * D_INNER + d) * D_STATE;
#pragma unroll
  for (int n = 0; n < D_STATE; ++n) s[n] = sinit[so + n];
  const float dsk = Dskip[d];

  const size_t base = (size_t)(b * SEQ + c * TC) * D_INNER + d;
  for (int tt = 0; tt < TC; ++tt) {
    const float dl = delta[base + (size_t)tt * D_INNER];
    const float uv = u[base + (size_t)tt * D_INNER];
    const float du = dl * uv;
    const float* Bt = &Bsh[tt * D_STATE];
    const float* Ct = &Csh[tt * D_STATE];
    float yacc = 0.f;
#pragma unroll
    for (int n = 0; n < D_STATE; ++n) {
      const float a = __expf(dl * Ad[n]);
      s[n] = fmaf(a, s[n], du * Bt[n]);
      yacc = fmaf(s[n], Ct[n], yacc);
    }
    yacc = fmaf(uv, dsk, yacc);
    const float zv = xz[(size_t)(b * SEQ + c * TC + tt) * (2 * D_INNER) + D_INNER + d];
    yh[base + (size_t)tt * D_INNER] = (_Float16)(yacc * silu_f(zv));
  }
}

// h += yh[16384,128] @ opw_h^T[128,64]; residual in C; writes f32 h + f16 hh
__global__ __launch_bounds__(128) void k_out_proj(
    const _Float16* __restrict__ yh, const _Float16* __restrict__ opw_h,
    float* __restrict__ h, _Float16* __restrict__ hh) {
  const int lane = threadIdx.x & 31, w = threadIdx.x >> 5;
  const int row0 = blockIdx.x * 16, n0 = w * 16;
  v8f acc = load_c(h, row0, n0, D_MODEL, lane);     // residual
#pragma unroll
  for (int k0 = 0; k0 < D_INNER; k0 += 32) {
    v16h a = load_frag(yh, row0, D_INNER, k0, lane);
    v16h b = load_frag(opw_h, n0, D_INNER, k0, lane);
    acc = wmma_f16(a, b, acc);
  }
  const int n = lane & 15, hhf = lane >> 4;
#pragma unroll
  for (int r = 0; r < 8; ++r) {
    const size_t o = (size_t)(row0 + r + 8 * hhf) * D_MODEL + n0 + n;
    h[o] = acc[r];
    hh[o] = (_Float16)acc[r];
  }
}

// mean over time + classifier head; one block per batch element
__global__ __launch_bounds__(256) void k_pool_cls(
    const float* __restrict__ h, const float* __restrict__ w_cls,
    const float* __restrict__ b_cls, float* __restrict__ out) {
  const int b = blockIdx.x, tid = threadIdx.x;
  const int m = tid & 63, q = tid >> 6;
  float s = 0.f;
  for (int t = q; t < SEQ; t += 4) s += h[(size_t)(b * SEQ + t) * D_MODEL + m];
  __shared__ float red[256];
  __shared__ float pooled[D_MODEL];
  red[tid] = s;
  __syncthreads();
  if (q == 0)
    pooled[m] = (red[m] + red[m + 64] + red[m + 128] + red[m + 192]) * (1.f / SEQ);
  __syncthreads();
  if (tid < OUT_SIZE) {
    float acc = b_cls[tid];
#pragma unroll
    for (int k = 0; k < D_MODEL; ++k) acc = fmaf(w_cls[tid * D_MODEL + k], pooled[k], acc);
    out[b * OUT_SIZE + tid] = acc;
  }
}

// ---------------- host launcher ----------------
extern "C" void kernel_launch(void* const* d_in, const int* in_sizes, int n_in,
                              void* d_out, int out_size, void* d_ws, size_t ws_size,
                              hipStream_t stream) {
  (void)in_sizes; (void)n_in; (void)out_size; (void)ws_size;
  const float* x         = (const float*)d_in[0];
  const float* w_in      = (const float*)d_in[1];
  const float* b_in      = (const float*)d_in[2];
  const float* in_proj_w = (const float*)d_in[3];
  const float* conv_w    = (const float*)d_in[4];
  const float* conv_b    = (const float*)d_in[5];
  const float* x_proj_w  = (const float*)d_in[6];
  const float* dt_proj_w = (const float*)d_in[7];
  const float* dt_proj_b = (const float*)d_in[8];
  const float* A_log     = (const float*)d_in[9];
  const float* D_skip    = (const float*)d_in[10];
  const float* out_proj_w= (const float*)d_in[11];
  const float* w_cls     = (const float*)d_in[12];
  const float* b_cls     = (const float*)d_in[13];
  float* out = (float*)d_out;

  // ---- workspace layout: f32 region then f16 region (~55 MB, L2-resident)
  float* W = (float*)d_ws;
  float* h     = W;                                // 16384*64
  float* xz    = h    + (size_t)ROWS * D_MODEL;    // 16384*256
  float* u     = xz   + (size_t)ROWS * 2 * D_INNER;
  float* delta = u    + (size_t)ROWS * D_INNER;
  float* Bm    = delta+ (size_t)ROWS * D_INNER;    // 16384*16
  float* Cm    = Bm   + (size_t)ROWS * D_STATE;
  float* send  = Cm   + (size_t)ROWS * D_STATE;    // 4*64*128*16
  float* aprod = send + (size_t)BATCH * NC * D_INNER * D_STATE;
  float* sinit = aprod+ (size_t)BATCH * NC * D_INNER * D_STATE;
  float* fend  = sinit+ (size_t)BATCH * NC * D_INNER * D_STATE;

  _Float16* H = (_Float16*)fend;                   // 16B-aligned (offset /16)
  _Float16* xh    = H;                             // 16384*64  (padded 57->64)
  _Float16* hh    = xh  + (size_t)ROWS * KPAD_IN;  // 16384*64
  _Float16* yh    = hh  + (size_t)ROWS * D_MODEL;  // 16384*128
  _Float16* w_in_h= yh  + (size_t)ROWS * D_INNER;  // 64*64 (padded)
  _Float16* ipw_h = w_in_h + (size_t)D_MODEL * KPAD_IN;       // 2*256*64
  _Float16* opw_h = ipw_h  + (size_t)N_LAYERS * 2 * D_INNER * D_MODEL; // 2*64*128

  const int MT = ROWS / 16;                        // 1024 M-tiles

  // ---- prep: pack all WMMA operands as padded f16
  {
    int n;
    n = ROWS * KPAD_IN;
    k_cvt_pad<<<(n + 255) / 256, 256, 0, stream>>>(xh, x, ROWS, INPUT_DIM, KPAD_IN);
    n = D_MODEL * KPAD_IN;
    k_cvt_pad<<<(n + 255) / 256, 256, 0, stream>>>(w_in_h, w_in, D_MODEL, INPUT_DIM, KPAD_IN);
    n = N_LAYERS * 2 * D_INNER * D_MODEL;
    k_cvt_pad<<<(n + 255) / 256, 256, 0, stream>>>(ipw_h, in_proj_w,
                                                   N_LAYERS * 2 * D_INNER, D_MODEL, D_MODEL);
    n = N_LAYERS * D_MODEL * D_INNER;
    k_cvt_pad<<<(n + 255) / 256, 256, 0, stream>>>(opw_h, out_proj_w,
                                                   N_LAYERS * D_MODEL, D_INNER, D_INNER);
  }

  k_linear_in<<<MT, 128, 0, stream>>>(xh, w_in_h, b_in, h, hh);

  for (int l = 0; l < N_LAYERS; ++l) {
    const _Float16* ipw = ipw_h + (size_t)l * (2 * D_INNER) * D_MODEL;
    const _Float16* opw = opw_h + (size_t)l * D_MODEL * D_INNER;
    const float* cw  = conv_w   + (size_t)l * D_INNER * D_CONV;
    const float* cb  = conv_b   + (size_t)l * D_INNER;
    const float* xpw = x_proj_w + (size_t)l * DBL_DIM * D_INNER;
    const float* dpw = dt_proj_w+ (size_t)l * D_INNER * DT_RANK;
    const float* dpb = dt_proj_b+ (size_t)l * D_INNER;
    const float* Al  = A_log    + (size_t)l * D_INNER * D_STATE;
    const float* Dp  = D_skip   + (size_t)l * D_INNER;

    k_in_proj<<<MT, 256, 0, stream>>>(hh, ipw, xz);
    k_mix<<<ROWS, 128, 0, stream>>>(xz, cw, cb, xpw, dpw, dpb, u, delta, Bm, Cm);
    k_scan1<<<dim3(NC, BATCH), 128, 0, stream>>>(delta, u, Bm, Al, send, aprod);
    k_combine<<<32, 256, 0, stream>>>(send, aprod, sinit);
    k_scan2<<<dim3(NC, BATCH), 128, 0, stream>>>(delta, u, Bm, Cm, Al, Dp, xz, sinit, yh);
    k_out_proj<<<MT, 128, 0, stream>>>(yh, opw, h, hh);
  }

  k_pool_cls<<<BATCH, 256, 0, stream>>>(h, w_cls, b_cls, out);
}
